// Head_82686710383014
// MI455X (gfx1250) — matrix operand order repaired and checked by reading, not compiled, not run on hardware
//
#include <hip/hip_runtime.h>
#include <hip/hip_bf16.h>

#define B_  4
#define T_  4096
#define C_  2048
#define HD_ 128

typedef __bf16 bf16;
typedef bf16  bf16x8  __attribute__((ext_vector_type(8)));
typedef bf16  bf16x16 __attribute__((ext_vector_type(16)));
typedef float v8f     __attribute__((ext_vector_type(8)));

__device__ __forceinline__ bf16 f2bf(float f) {
  return (bf16)f;                       // native fptrunc -> v_cvt_pk_bf16_f32 (RNE)
}

__device__ __forceinline__ bf16x16 load_bf16x16(const bf16* p) {
  // 8 bf16 at p, 8 bf16 at p+16 (K and K+16 runs), 16B-aligned
  bf16x8 lo = *(const bf16x8*)p;
  bf16x8 hi = *(const bf16x8*)(p + 16);
  bf16x16 r;
  #pragma unroll
  for (int i = 0; i < 8; ++i) { r[i] = lo[i]; r[i + 8] = hi[i]; }
  return r;
}

__device__ __forceinline__ bf16x16 load_a_frag(const bf16* p) {
  // A-layout per lane: 8 bf16 at p, 8 bf16 at p+16 (K and K+16 runs)
  bf16x8 lo = *(const bf16x8*)p;
  bf16x8 hi = *(const bf16x8*)(p + 16);
  bf16x16 r;
  #pragma unroll
  for (int i = 0; i < 8; ++i) { r[i] = lo[i]; r[i + 8] = hi[i]; }
  return r;
}

__device__ __forceinline__ v8f wmma_bf16(bf16x16 a, bf16x16 b, v8f c) {
  return __builtin_amdgcn_wmma_f32_16x16x32_bf16(
      false, a, false, b, (short)0, c, false, false);
}

// ---------------------------------------------------------------------------
// Kernel 1: fp32 weights -> bf16. Wq gets log2(e)/sqrt(C) folded in so the
// flash softmax can use raw v_exp_f32 (base-2) without per-element muls.
// ---------------------------------------------------------------------------
__global__ void __launch_bounds__(256)
wconv_kernel(const float* __restrict__ Wq, const float* __restrict__ Wk,
             const float* __restrict__ Wv, bf16* __restrict__ wb)
{
  const int i   = blockIdx.x * 256 + threadIdx.x;
  const int per = HD_ * C_;
  float v;
  if (i < per)          v = Wq[i] * (0.022097086912079608f * 1.4426950408889634f);
  else if (i < 2 * per) v = Wk[i - per];
  else                  v = Wv[i - 2 * per];
  wb[i] = f2bf(v);
}

// ---------------------------------------------------------------------------
// Kernel 2: QKV projection + RoPE. 3 waves/block; wave w computes projection w
// for the same 16-row tile of x. K loop is register double-buffered: loads for
// step k+32 run while WMMAs consume step k. RoPE via __shfl_xor(.,1).
// ---------------------------------------------------------------------------
__global__ void __launch_bounds__(96)
qkv_rope_kernel(const float* __restrict__ x,  const bf16* __restrict__ wb,
                const float* __restrict__ cosb, const float* __restrict__ sinb,
                bf16* __restrict__ qb, bf16* __restrict__ kb, bf16* __restrict__ vb)
{
  const int wave = threadIdx.x >> 5;          // 0=q, 1=k, 2=v
  const int lane = threadIdx.x & 31;
  const int lh   = lane & 15;
  const int hi   = lane >> 4;
  const long row0 = (long)blockIdx.x * 16;    // flattened (B*T) row base

  const bf16* W = wb + (size_t)wave * (HD_ * C_);
  const float* xrow = x + (row0 + lh) * (long)C_;
  const int akb = hi * 8;                     // A K-run base per lane half
  const int bkb = hi * 8;                     // B K-run base per lane half

  auto buildA = [&](int k0) {
    bf16x16 a;
    const float* p = xrow + k0 + akb;
    #pragma unroll
    for (int i = 0; i < 8; ++i) a[i]     = f2bf(p[i]);
    #pragma unroll
    for (int i = 0; i < 8; ++i) a[i + 8] = f2bf(p[16 + i]);
    return a;
  };
  auto loadB = [&](int k0, bf16x16* dst) {
    #pragma unroll
    for (int nb = 0; nb < 8; ++nb)
      dst[nb] = load_bf16x16(W + (size_t)(nb * 16 + lh) * C_ + k0 + bkb);
  };

  v8f acc[8];
  #pragma unroll
  for (int i = 0; i < 8; ++i) acc[i] = {};

  bf16x16 aA, bA[8];
  aA = buildA(0);
  loadB(0, bA);

  for (int k0 = 0; k0 < C_; k0 += 64) {
    // fill buffer B for step k0+32 while WMMAs consume buffer A
    bf16x16 aB, bB[8];
    aB = buildA(k0 + 32);
    loadB(k0 + 32, bB);
    if (k0 + akb + 256 < C_)
      __builtin_prefetch(xrow + k0 + akb + 256, 0, 3);   // x is the HBM stream
    #pragma unroll
    for (int nb = 0; nb < 8; ++nb)
      acc[nb] = wmma_bf16(aA, bA[nb], acc[nb]);

    // refill buffer A for step k0+64 while WMMAs consume buffer B
    if (k0 + 64 < C_) {
      aA = buildA(k0 + 64);
      loadB(k0 + 64, bA);
    }
    #pragma unroll
    for (int nb = 0; nb < 8; ++nb)
      acc[nb] = wmma_bf16(aB, bB[nb], acc[nb]);
  }

  bf16* outp = (wave == 0) ? qb : (wave == 1) ? kb : vb;
  const bool do_rope = (wave < 2);
  #pragma unroll
  for (int nb = 0; nb < 8; ++nb) {
    const int n = nb * 16 + lh;
    #pragma unroll
    for (int r = 0; r < 8; ++r) {
      const long grow = row0 + r + 8 * hi;    // C-layout: M = r + 8*hi
      float v = acc[nb][r];
      float other = __shfl_xor(v, 1, 32);     // RoPE partner (n^1), wave-uniform path
      float res = v;
      if (do_rope) {
        const int t = (int)(grow & (T_ - 1));
        const float c = cosb[t * (HD_ / 2) + (n >> 1)];
        const float s = sinb[t * (HD_ / 2) + (n >> 1)];
        res = (n & 1) ? __builtin_fmaf(other, s, v * c)     // vr*s + vi*c
                      : __builtin_fmaf(v, c, -(other * s)); // vr*c - vi*s
      }
      outp[grow * HD_ + n] = f2bf(res);
    }
  }
}

// ---------------------------------------------------------------------------
// Kernel 3: causal flash attention. 2 waves x 16 query rows (32-row blocks ->
// 512 workgroups for occupancy), 32 keys/step. K B-fragments load straight
// from global; V staged transposed in LDS; P round-trips a per-wave LDS tile.
// Softmax runs in the log2 domain (scores pre-scaled by log2(e)/sqrt(C)).
// ---------------------------------------------------------------------------
__global__ void __launch_bounds__(64)
flash_kernel(const bf16* __restrict__ qb, const bf16* __restrict__ kb,
             const bf16* __restrict__ vb, float* __restrict__ out)
{
  const int lane = threadIdx.x & 31;
  const int wave = threadIdx.x >> 5;          // 0..1
  const int lh   = lane & 15;
  const int hi   = lane >> 4;
  const int b    = blockIdx.x >> 7;           // T/32 = 128 row blocks per batch
  const int m0wg = (blockIdx.x & 127) * 32;
  const long base = (long)b * T_;
  const int m0 = m0wg + wave * 16;

  __shared__ bf16 Vt[HD_ * 32];               // V transposed: [d][s], 8 KB
  __shared__ bf16 Pb[2][16 * 32];             // per-wave P tile, 2 KB

  // Q fragments: 4 chunks of 16x32, resident for the whole loop
  bf16x16 qf[4];
  {
    const bf16* qrow = qb + (base + m0 + lh) * HD_;
    #pragma unroll
    for (int c = 0; c < 4; ++c)
      qf[c] = load_a_frag(qrow + c * 32 + hi * 8);
  }

  v8f o[8];
  #pragma unroll
  for (int i = 0; i < 8; ++i) o[i] = {};
  float mrow[8], lrow[8];
  #pragma unroll
  for (int r = 0; r < 8; ++r) { mrow[r] = -__builtin_inff(); lrow[r] = 0.f; }

  const int sEnd = m0wg + 32;
  for (int s0 = 0; s0 < sEnd; s0 += 32) {
    // --- cooperative V block load + transpose into LDS (64 threads) ---
    #pragma unroll
    for (int kk = 0; kk < 8; ++kk) {
      const int c  = threadIdx.x + kk * 64;   // 0..511 chunks of 8
      const int s  = c >> 4;                  // 0..31
      const int d0 = (c & 15) * 8;
      const bf16* vp = vb + (base + s0 + s) * HD_ + d0;
      bf16x8 vv = *(const bf16x8*)vp;
      if (s0 + 32 < sEnd)
        __builtin_prefetch(vp + 32 * HD_, 0, 3);   // next key block
      #pragma unroll
      for (int e = 0; e < 8; ++e) Vt[(d0 + e) * 32 + s] = vv[e];
    }
    __syncthreads();

    if (s0 <= m0) {                           // causal: skip fully-masked blocks
      // ---- S = Q * K^T: preload all 8 K fragments, then 8 WMMAs ----
      v8f sacc[2]; sacc[0] = {}; sacc[1] = {};
      bf16x16 kf[8];
      #pragma unroll
      for (int t = 0; t < 2; ++t) {
        const bf16* krow = kb + (base + s0 + t * 16 + lh) * HD_ + hi * 8;
        #pragma unroll
        for (int c = 0; c < 4; ++c) kf[t * 4 + c] = load_bf16x16(krow + c * 32);
      }
      #pragma unroll
      for (int t = 0; t < 2; ++t)
        #pragma unroll
        for (int c = 0; c < 4; ++c)
          sacc[t] = wmma_bf16(qf[c], kf[t * 4 + c], sacc[t]);

      // ---- causal mask on diagonal-overlapping blocks ----
      if (s0 + 31 > m0) {
        #pragma unroll
        for (int t = 0; t < 2; ++t) {
          const int scol = s0 + t * 16 + lh;
          #pragma unroll
          for (int r = 0; r < 8; ++r)
            if (scol > m0 + r + 8 * hi) sacc[t][r] = -__builtin_inff();
        }
      }
      // ---- online softmax in log2 domain, rows within each 16-lane half ----
      float alpha[8];
      #pragma unroll
      for (int r = 0; r < 8; ++r) {
        float mx = fmaxf(sacc[0][r], sacc[1][r]);
        mx = fmaxf(mx, __shfl_xor(mx, 1, 32));
        mx = fmaxf(mx, __shfl_xor(mx, 2, 32));
        mx = fmaxf(mx, __shfl_xor(mx, 4, 32));
        mx = fmaxf(mx, __shfl_xor(mx, 8, 32));
        const float mnew = fmaxf(mrow[r], mx);
        const float a  = __builtin_amdgcn_exp2f(mrow[r] - mnew);
        const float p0 = __builtin_amdgcn_exp2f(sacc[0][r] - mnew);
        const float p1 = __builtin_amdgcn_exp2f(sacc[1][r] - mnew);
        sacc[0][r] = p0; sacc[1][r] = p1;
        float ps = p0 + p1;
        ps += __shfl_xor(ps, 1, 32);
        ps += __shfl_xor(ps, 2, 32);
        ps += __shfl_xor(ps, 4, 32);
        ps += __shfl_xor(ps, 8, 32);
        lrow[r] = lrow[r] * a + ps;
        mrow[r] = mnew;
        alpha[r] = a;
      }
      #pragma unroll
      for (int db = 0; db < 8; ++db)
        #pragma unroll
        for (int r = 0; r < 8; ++r) o[db][r] *= alpha[r];

      // ---- P (C-layout f32) -> per-wave LDS -> bf16 A fragment ----
      bf16* P = Pb[wave];
      #pragma unroll
      for (int t = 0; t < 2; ++t)
        #pragma unroll
        for (int r = 0; r < 8; ++r)
          P[(r + 8 * hi) * 32 + t * 16 + lh] = f2bf(sacc[t][r]);
      bf16x16 pf = load_a_frag(P + lh * 32 + hi * 8);

      // ---- O += P * V (8 d-blocks, B fragments from transposed LDS) ----
      #pragma unroll
      for (int g = 0; g < 2; ++g) {
        bf16x16 vf[4];
        #pragma unroll
        for (int j = 0; j < 4; ++j) {
          const int db = g * 4 + j;
          vf[j] = load_bf16x16(&Vt[(db * 16 + lh) * 32 + hi * 8]);
        }
        #pragma unroll
        for (int j = 0; j < 4; ++j)
          o[g * 4 + j] = wmma_bf16(pf, vf[j], o[g * 4 + j]);
      }
    }
    __syncthreads();
  }

  // ---- epilogue: O / l, fp32 out ----
  float* op = out + (base + m0) * (long)HD_;
  #pragma unroll
  for (int r = 0; r < 8; ++r) {
    const float inv = 1.0f / lrow[r];
    const int m = r + 8 * hi;
    #pragma unroll
    for (int db = 0; db < 8; ++db)
      op[(long)m * HD_ + db * 16 + lh] = o[db][r] * inv;
  }
}

// ---------------------------------------------------------------------------
extern "C" void kernel_launch(void* const* d_in, const int* in_sizes, int n_in,
                              void* d_out, int out_size, void* d_ws, size_t ws_size,
                              hipStream_t stream)
{
  (void)in_sizes; (void)n_in; (void)out_size; (void)ws_size;
  const float* x  = (const float*)d_in[0];
  const float* Wq = (const float*)d_in[1];
  const float* Wk = (const float*)d_in[2];
  const float* Wv = (const float*)d_in[3];
  const float* cs = (const float*)d_in[4];
  const float* sn = (const float*)d_in[5];
  float* out = (float*)d_out;

  char* ws = (char*)d_ws;
  bf16* wb = (bf16*)ws;                                   // 1.5 MB bf16 weights
  bf16* qbuf = (bf16*)(ws + (size_t)3 * HD_ * C_ * 2);    // 4 MB each
  bf16* kbuf = qbuf + (size_t)B_ * T_ * HD_;
  bf16* vbuf = kbuf + (size_t)B_ * T_ * HD_;

  wconv_kernel<<<(3 * HD_ * C_) / 256, 256, 0, stream>>>(Wq, Wk, Wv, wb);
  qkv_rope_kernel<<<(B_ * T_) / 16, 96, 0, stream>>>(x, wb, cs, sn, qbuf, kbuf, vbuf);
  flash_kernel<<<B_ * (T_ / 32), 64, 0, stream>>>(qbuf, kbuf, vbuf, out);
}
